// GRUmodule_43817256354350
// MI455X (gfx1250) — matrix operand order repaired and checked by reading, not compiled, not run on hardware
//
#include <hip/hip_runtime.h>
#include <hip/hip_bf16.h>

typedef __attribute__((ext_vector_type(16))) __bf16 v16bf;
typedef __attribute__((ext_vector_type(8)))  __bf16 v8bf;
typedef __attribute__((ext_vector_type(8)))  float  v8f;

#define B_  128
#define S_  1024
#define F_  256
#define H_  512
#define K_  (F_ + H_)     // 768
#define KT_ 24            // K_/32 k-tiles
#define JT_ 32            // H_/16 n-tiles per gate
#define MT_ 8             // B_/16 m-tiles
#define LDA 776           // padded LDS row stride (elems); 776*2 % 16 == 0, bank-skewed

// ---------------------------------------------------------------------------
// Pack the six weight matrices into WMMA-native B fragments (bf16).
// Fragment (g, j, kt): 32 lanes x 16 bf16 contiguous; lane L holds
// N = j*16 + (L&15), K = kt*32 + (L>>4)*8 + (e&7) + (e>>3)*16   (CDNA5 ISA B layout).
// Gate 0: [W_iu;W_hu], 1: [W_ir;W_hr], 2: W_io (K<256), 3: W_ho (K>=256).
// ---------------------------------------------------------------------------
__global__ __launch_bounds__(256) void gru_pack_b(
    const float* __restrict__ W_iu, const float* __restrict__ W_hu,
    const float* __restrict__ W_ir, const float* __restrict__ W_hr,
    const float* __restrict__ W_io, const float* __restrict__ W_ho,
    __bf16* __restrict__ Bp)
{
    int idx = blockIdx.x * 256 + threadIdx.x;   // 4*32*24*512 = 1,572,864 total
    int e    = idx & 15;
    int L    = (idx >> 4) & 31;
    int frag = idx >> 9;
    int kt   = frag % KT_;
    int j    = (frag / KT_) % JT_;
    int g    = frag / (KT_ * JT_);

    int n = j * 16 + (L & 15);
    int k = kt * 32 + (L >> 4) * 8 + (e & 7) + ((e >> 3) << 4);

    float v = 0.0f;
    if (g == 0)      v = (k < F_) ? W_iu[k * H_ + n] : W_hu[(k - F_) * H_ + n];
    else if (g == 1) v = (k < F_) ? W_ir[k * H_ + n] : W_hr[(k - F_) * H_ + n];
    else if (g == 2) { if (k <  F_) v = W_io[k * H_ + n]; }
    else             { if (k >= F_) v = W_ho[(k - F_) * H_ + n]; }
    Bp[idx] = (__bf16)v;
}

__global__ __launch_bounds__(256) void gru_zero(float* __restrict__ p, int n) {
    int i = blockIdx.x * 256 + threadIdx.x;
    if (i < n) p[i] = 0.0f;
}

// ---------------------------------------------------------------------------
// Compile-time-bounded WMMA accumulation over k-tiles [K0, K1).
// Two interleaved accumulators break the WMMA D->C dependency chain.
// ---------------------------------------------------------------------------
template <int K0, int K1>
__device__ __forceinline__ v8f accum_tiles(const __bf16* __restrict__ aRow,
                                           const __bf16* __restrict__ bp,
                                           int lane, int half)
{
    v8f acc0 = {};
    v8f acc1 = {};
#pragma unroll
    for (int kt = K0; kt < K1; ++kt) {
        // A fragment: two 16B LDS loads (K = k0..k0+7 and k0+16..k0+23)
        const v8bf* ap = (const v8bf*)(aRow + kt * 32 + half * 8);
        v8bf alo = ap[0];
        v8bf ahi = ap[2];               // +16 elements
        v16bf a;
#pragma unroll
        for (int q = 0; q < 8; ++q) { a[q] = alo[q]; a[q + 8] = ahi[q]; }

        // B fragment: one 32B coalesced global load per lane (L2-resident)
        v16bf b = *(const v16bf*)(bp + (((size_t)kt) << 9) + (lane << 4));

        if ((kt - K0) & 1)
            acc1 = __builtin_amdgcn_wmma_f32_16x16x32_bf16(
                       false, a, false, b, (short)0, acc1, false, false);
        else
            acc0 = __builtin_amdgcn_wmma_f32_16x16x32_bf16(
                       false, a, false, b, (short)0, acc0, false, false);
    }
    return acc0 + acc1;
}

__device__ __forceinline__ float fast_sigmoid(float x) {
    return __builtin_amdgcn_rcpf(1.0f + __expf(-x));
}

__device__ __forceinline__ float fast_tanh(float x) {
#if __has_builtin(__builtin_amdgcn_tanhf)
    return __builtin_amdgcn_tanhf(x);
#else
    float e2 = __expf(2.0f * x);
    return (e2 - 1.0f) * __builtin_amdgcn_rcpf(e2 + 1.0f);
#endif
}

// ---------------------------------------------------------------------------
// One recurrence step. Block = 4 waves; block (j, mb) owns batch rows
// [mb*16, mb*16+16) and hidden cols [j*16, j*16+16).
// Wave w computes gate-w's 16x16 pre-activation tile via WMMA over the
// LDS-staged A = [x_t | h] (16 x 768 bf16), then the block fuses the
// sigmoid/tanh gates and the h update.
// ---------------------------------------------------------------------------
__global__ __launch_bounds__(128) void gru_step(
    const float*  __restrict__ x,     // [B,S,F] f32
    const float*  __restrict__ h_in,  // [B,H]  f32
    float*        __restrict__ h_out, // [B,H]  f32
    const __bf16* __restrict__ Bp,    // packed B fragments
    const float*  __restrict__ b_u,
    const float*  __restrict__ b_r,
    const float*  __restrict__ b_o,
    int t)
{
    __shared__ __attribute__((aligned(16))) __bf16 sA[16 * LDA];
    __shared__ float sG[4 * 16 * 17];

    const int tid    = threadIdx.x;
    const int j      = blockIdx.x;      // 0..31 hidden n-tile
    const int mb     = blockIdx.y;      // 0..7  batch m-tile
    const int m_base = mb * 16;

    // ---- Stage A = [x_t | h] rows m_base..+15, f32 -> bf16, coalesced ----
#pragma unroll 4
    for (int row = 0; row < 16; ++row) {
        const int b = m_base + row;
        const float* xr = x + ((size_t)b * S_ + t) * F_;
        const float* hr = h_in + (size_t)b * H_;
        __bf16* dr = sA + row * LDA;
#pragma unroll
        for (int p = 0; p < 2; ++p) {        // cols [0,256): x_t
            int col = p * 128 + tid;
            dr[col] = (__bf16)xr[col];
        }
#pragma unroll
        for (int p = 0; p < 4; ++p) {        // cols [256,768): h
            int col = p * 128 + tid;
            dr[F_ + col] = (__bf16)hr[col];
        }
    }
    __syncthreads();

    const int lane = tid & 31;
    const int lrow = lane & 15;         // A: M row / B: N col / D: N col
    const int half = lane >> 4;
    const int w    = __builtin_amdgcn_readfirstlane(tid >> 5);  // gate id (scalar)

    const __bf16* aRow = sA + lrow * LDA;
    const __bf16* bp   = Bp + ((size_t)((w * JT_ + j) * KT_) << 9);

    v8f acc;
    if (w == 2)      acc = accum_tiles<0, 8 >(aRow, bp, lane, half);  // o_x: K [0,256)
    else if (w == 3) acc = accum_tiles<8, 24>(aRow, bp, lane, half);  // o_h: K [256,768)
    else             acc = accum_tiles<0, 24>(aRow, bp, lane, half);  // u, r: full K

    // ---- Park gate tile in LDS (D layout: VGPR r -> M = r + 8*half) ----
#pragma unroll
    for (int r8 = 0; r8 < 8; ++r8) {
        int M = r8 + half * 8;
        sG[w * 272 + M * 17 + lrow] = acc[r8];
    }
    __syncthreads();

    // ---- Fused gate math + h update: 256 outputs, 2 per thread ----
#pragma unroll
    for (int i = tid; i < 256; i += 128) {
        int m  = i >> 4;
        int n  = i & 15;
        int b  = m_base + m;
        int jj = j * 16 + n;

        float gu = sG[0 * 272 + m * 17 + n] + b_u[jj];
        float gr = sG[1 * 272 + m * 17 + n] + b_r[jj];
        float gx = sG[2 * 272 + m * 17 + n];
        float gh = sG[3 * 272 + m * 17 + n];

        float u = fast_sigmoid(gu);
        float r = fast_sigmoid(gr);
        float o = fast_tanh(gx + r * gh + b_o[jj]);

        float hv = h_in[b * H_ + jj];
        h_out[b * H_ + jj] = (1.0f - u) * hv + u * o;
    }
}

// ---------------------------------------------------------------------------
extern "C" void kernel_launch(void* const* d_in, const int* in_sizes, int n_in,
                              void* d_out, int out_size, void* d_ws, size_t ws_size,
                              hipStream_t stream)
{
    (void)in_sizes; (void)n_in; (void)out_size; (void)ws_size;

    const float* x    = (const float*)d_in[0];
    const float* W_iu = (const float*)d_in[1];
    const float* W_hu = (const float*)d_in[2];
    const float* b_u  = (const float*)d_in[3];
    const float* W_ir = (const float*)d_in[4];
    const float* W_hr = (const float*)d_in[5];
    const float* b_r  = (const float*)d_in[6];
    const float* W_io = (const float*)d_in[7];
    const float* W_ho = (const float*)d_in[8];
    const float* b_o  = (const float*)d_in[9];

    const size_t bp_elems = (size_t)4 * JT_ * KT_ * 512;   // 1,572,864 bf16
    __bf16* Bp = (__bf16*)d_ws;
    float*  h0 = (float*)((char*)d_ws + bp_elems * sizeof(__bf16));
    float*  h1 = h0 + (size_t)B_ * H_;

    gru_pack_b<<<(int)(bp_elems / 256), 256, 0, stream>>>(
        W_iu, W_hu, W_ir, W_hr, W_io, W_ho, Bp);
    gru_zero<<<(B_ * H_ + 255) / 256, 256, 0, stream>>>(h0, B_ * H_);

    dim3 grid(JT_, MT_);   // 32 x 8 = 256 blocks, 128 threads each
    for (int t = 0; t < S_; ++t) {
        float* hin  = (t & 1) ? h1 : h0;
        float* hout = (t & 1) ? h0 : h1;
        gru_step<<<grid, 128, 0, stream>>>(x, hin, hout, Bp, b_u, b_r, b_o, t);
    }
    // S_ = 1024 (even) -> final state lands back in h0
    hipMemcpyAsync(d_out, h0, (size_t)B_ * H_ * sizeof(float),
                   hipMemcpyDeviceToDevice, stream);
}
